// HybridGATLSTM_34815004902110
// MI455X (gfx1250) — matrix operand order repaired and checked by reading, not compiled, not run on hardware
//
#include <hip/hip_runtime.h>
#include <hip/hip_bf16.h>
#include <stdint.h>

// ---------------------------------------------------------------------------
// HybridGATLSTM for MI455X (gfx1250, wave32, WMMA).
//  - BiLSTM layers via v_wmma_f32_16x16x32_f16, one 16-node tile per wave.
//  - Weights pre-packed into B-fragment order; activations written in
//    A-fragment order so every WMMA operand is a contiguous 32B/lane load.
//  - Gate nonlinearities use HW transcendentals (v_tanh_f32 / v_exp_f32) so
//    they co-execute as TRANS ops next to the WMMA stream.
//  - GAT softmax aggregation via 3 atomic scatter passes (L2-resident).
// ---------------------------------------------------------------------------

typedef __attribute__((ext_vector_type(16))) _Float16 v16h;
typedef __attribute__((ext_vector_type(8)))  float    v8f;

#define NNODES 100000
#define NEDGES 3200000
#define TSEQ   5
#define NTILES 6250         // NNODES / 16 (exact)
#define HEADS  6
#define FDIM   24           // HEADS*4

#define LOG2E 1.44269504088896340736f

// ---- fast hardware transcendentals (guarded; fall back to libm) -----------
__device__ __forceinline__ float fast_exp(float x){
#if __has_builtin(__builtin_amdgcn_exp2f)
  return __builtin_amdgcn_exp2f(x * LOG2E);
#else
  return __expf(x);
#endif
}

__device__ __forceinline__ float fast_rcp(float x){
#if __has_builtin(__builtin_amdgcn_rcpf)
  return __builtin_amdgcn_rcpf(x);
#else
  return 1.f/x;
#endif
}

__device__ __forceinline__ float fast_sig(float x){
  // 1/(1+exp(-x)) : v_exp_f32 + v_add + v_rcp_f32
  return fast_rcp(1.f + fast_exp(-x));
}

__device__ __forceinline__ float fast_tanh(float x){
#if __has_builtin(__builtin_amdgcn_tanhf)
  return __builtin_amdgcn_tanhf(x);          // v_tanh_f32 (gfx1250 TRANS op)
#elif __has_builtin(__builtin_amdgcn_tanh_f32)
  return __builtin_amdgcn_tanh_f32(x);
#else
  float e = fast_exp(2.f*x);                 // (e-1)/(e+1)
  return 1.f - 2.f*fast_rcp(e + 1.f);
#endif
}

__device__ __forceinline__ v8f wmma16(v16h a, v16h b, v8f c){
  // D = A(16x32 f16) * B(32x16 f16) + C(16x16 f32)
  return __builtin_amdgcn_wmma_f32_16x16x32_f16(false, a, false, b, (short)0, c,
                                                false, false);
}

// Gather a 16x32 f16 A-fragment from a row-major [16][rowstride] LDS tile.
// A layout: lanes 0-15 rows M=lane, lanes 16-31 same rows with K offset +8;
// half i -> k = i + (i>=8 ? 8 : 0) + (lane>=16 ? 8 : 0).
__device__ __forceinline__ v16h load_afrag(const _Float16* st, int rowstride,
                                           int kbase, int lane){
  int m    = lane & 15;
  int lhi8 = (lane >> 4) << 3;
  v16h a;
  #pragma unroll
  for (int i = 0; i < 16; ++i){
    int k = i + ((i >> 3) << 3) + lhi8;
    a[i] = st[m*rowstride + kbase + k];
  }
  return a;
}

// Store one element (row m, global-k kg) into an A-fragment-ordered buffer
// laid out as [KB][32 lanes][16 halves].
__device__ __forceinline__ void store_afrag_elem(_Float16* base, int kg, int m,
                                                 float v){
  int kb  = kg >> 5, k32 = kg & 31;
  int lhi = (k32 >> 3) & 1;
  int ii  = (k32 & 7) + ((k32 >> 4) << 3);
  base[kb*512 + (m + 16*lhi)*16 + ii] = (_Float16)v;
}

// Ordered-uint encoding so atomicMax on u32 == float max.
__device__ __forceinline__ unsigned fenc(float f){
  unsigned u = __float_as_uint(f);
  return (u & 0x80000000u) ? ~u : (u | 0x80000000u);
}
__device__ __forceinline__ float fdec(unsigned u){
  unsigned v = (u & 0x80000000u) ? (u ^ 0x80000000u) : ~u;
  return __uint_as_float(v);
}

// ---------------------------------------------------------------------------
// Pack a weight matrix (f32) into f16 B-fragment order: frag = kb*NB + nb,
// entry [frag][lane][i] = B[k][n] with n = nb*16 + (lane&15),
// k = kb*32 + (i>=8 ? 16 : 0) + (lane>=16 ? 8 : 0) + (i&7).
// src_is_wT = 1: src is (Nv x Kv) row-major and B = src^T (gates = h @ w^T).
// src_is_wT = 0: src is (Kv x Nv) row-major and B = src    (h @ W).
// ---------------------------------------------------------------------------
__global__ void pack_bfrag_k(const float* __restrict__ src,
                             _Float16* __restrict__ dst,
                             int NB, int KB, int Nv, int Kv, int src_is_wT){
  int idx = blockIdx.x*blockDim.x + threadIdx.x;
  if (idx >= NB*KB*512) return;
  int frag = idx >> 9;
  int l    = (idx >> 4) & 31;
  int i    = idx & 15;
  int kb = frag / NB, nb = frag - kb*NB;
  int n = nb*16 + (l & 15);
  int k = kb*32 + ((i >> 3) << 4) + ((l >> 4) << 3) + (i & 7);
  float v = 0.f;
  if (n < Nv && k < Kv) v = src_is_wT ? src[n*Kv + k] : src[k*Nv + n];
  dst[idx] = (_Float16)v;
}

// ---------------------------------------------------------------------------
// LSTM layer 0 (input dim 1), both directions. One wave per 16-node tile.
// Output h0frag: per (tile,t), 4 A-fragments of the 16x128 concat(f,b) row.
// ---------------------------------------------------------------------------
__global__ void __launch_bounds__(256)
lstm0_k(const float* __restrict__ x,
        const float* __restrict__ wih_f, const float* __restrict__ wih_b,
        const float* __restrict__ bih_f, const float* __restrict__ bhh_f,
        const float* __restrict__ bih_b, const float* __restrict__ bhh_b,
        const _Float16* __restrict__ whhfrag_f,
        const _Float16* __restrict__ whhfrag_b,
        _Float16* __restrict__ h0frag){
  __shared__ _Float16 hstage[8][16*64];
  __shared__ float    xs[8][16][TSEQ];
  int w = threadIdx.x >> 5, lane = threadIdx.x & 31;
  int tile = blockIdx.x*8 + w;
  if (tile >= NTILES) return;
  _Float16* st = hstage[w];
  int base = tile*16;
  for (int idx = lane; idx < 16*TSEQ; idx += 32){
    int m = idx/TSEQ, t = idx - m*TSEQ;
    xs[w][m][t] = x[(size_t)(base+m)*TSEQ + t];
  }
  int mrow_hi = (lane >> 4) << 3;
  int ncol    = lane & 15;

  #pragma unroll 1
  for (int dir = 0; dir < 2; ++dir){
    const float*    wih  = dir ? wih_b : wih_f;
    const float*    bih  = dir ? bih_b : bih_f;
    const float*    bhh  = dir ? bhh_b : bhh_f;
    const _Float16* whhf = dir ? whhfrag_b : whhfrag_f;

    for (int i = lane; i < 512; i += 32) ((unsigned*)st)[i] = 0u;
    float c_st[4][8];
    #pragma unroll
    for (int j = 0; j < 4; ++j)
      #pragma unroll
      for (int r = 0; r < 8; ++r) c_st[j][r] = 0.f;
    asm volatile("s_wait_dscnt 0" ::: "memory");

    #pragma unroll 1
    for (int s = 0; s < TSEQ; ++s){
      int t = dir ? (TSEQ-1-s) : s;
      v16h a0 = load_afrag(st, 64, 0,  lane);
      v16h a1 = load_afrag(st, 64, 32, lane);
      _Float16* obase = h0frag + (size_t)(tile*TSEQ + t)*2048;

      #pragma unroll
      for (int j = 0; j < 4; ++j){
        v8f acc[4];
        #pragma unroll
        for (int g = 0; g < 4; ++g){
          int col = g*64 + j*16 + ncol;
          float binit = bih[col] + bhh[col];
          float wv    = wih[col];
          #pragma unroll
          for (int r = 0; r < 8; ++r)
            acc[g][r] = binit + xs[w][r + mrow_hi][t]*wv;
        }
        #pragma unroll
        for (int g = 0; g < 4; ++g){
          int nb = g*4 + j;
          v16h b0 = *(const v16h*)(whhf + (size_t)(nb)*512      + lane*16);
          acc[g] = wmma16(a0, b0, acc[g]);
          v16h b1 = *(const v16h*)(whhf + (size_t)(16 + nb)*512 + lane*16);
          acc[g] = wmma16(a1, b1, acc[g]);
        }
        #pragma unroll
        for (int r = 0; r < 8; ++r){
          float iv = fast_sig(acc[0][r]);
          float fv = fast_sig(acc[1][r]);
          float gv = fast_tanh(acc[2][r]);
          float ov = fast_sig(acc[3][r]);
          float cv = fv*c_st[j][r] + iv*gv;
          c_st[j][r] = cv;
          float hv = ov*fast_tanh(cv);
          int m = r + mrow_hi;
          int n = j*16 + ncol;
          st[m*64 + n] = (_Float16)hv;
          store_afrag_elem(obase, dir*64 + n, m, hv);
        }
      }
      asm volatile("s_wait_dscnt 0" ::: "memory");
    }
  }
}

// ---------------------------------------------------------------------------
// LSTM layer 1 (input dim 128). Forward full sequence; backward is exactly
// one step at t=T-1 (reverse scan's last emitted state) with h0=c0=0.
// Only the final [:, -1, :] row is needed -> hfinfrag (16x128 per tile).
// ---------------------------------------------------------------------------
__global__ void __launch_bounds__(256)
lstm1_k(const _Float16* __restrict__ h0frag,
        const float* __restrict__ bih_f, const float* __restrict__ bhh_f,
        const float* __restrict__ bih_b, const float* __restrict__ bhh_b,
        const _Float16* __restrict__ wihfrag_f,
        const _Float16* __restrict__ whhfrag_f,
        const _Float16* __restrict__ wihfrag_b,
        _Float16* __restrict__ hfinfrag){
  __shared__ _Float16 hstage[8][16*64];
  int w = threadIdx.x >> 5, lane = threadIdx.x & 31;
  int tile = blockIdx.x*8 + w;
  if (tile >= NTILES) return;
  _Float16* st = hstage[w];
  int mrow_hi = (lane >> 4) << 3;
  int ncol    = lane & 15;
  _Float16* obase = hfinfrag + (size_t)tile*2048;

  // ---- forward direction ----
  for (int i = lane; i < 512; i += 32) ((unsigned*)st)[i] = 0u;
  float c_st[4][8];
  #pragma unroll
  for (int j = 0; j < 4; ++j)
    #pragma unroll
    for (int r = 0; r < 8; ++r) c_st[j][r] = 0.f;
  asm volatile("s_wait_dscnt 0" ::: "memory");

  #pragma unroll 1
  for (int t = 0; t < TSEQ; ++t){
    v16h ain[4];
    #pragma unroll
    for (int kb = 0; kb < 4; ++kb)
      ain[kb] = *(const v16h*)(h0frag + ((size_t)(tile*TSEQ + t)*4 + kb)*512
                               + lane*16);
    if (t + 1 < TSEQ)
      __builtin_prefetch(h0frag + ((size_t)(tile*TSEQ + t + 1)*4)*512 + lane*16,
                         0, 3);
    v16h ah0 = load_afrag(st, 64, 0,  lane);
    v16h ah1 = load_afrag(st, 64, 32, lane);

    #pragma unroll
    for (int j = 0; j < 4; ++j){
      v8f acc[4];
      #pragma unroll
      for (int g = 0; g < 4; ++g){
        int col = g*64 + j*16 + ncol;
        float binit = bih_f[col] + bhh_f[col];
        #pragma unroll
        for (int r = 0; r < 8; ++r) acc[g][r] = binit;
      }
      #pragma unroll
      for (int g = 0; g < 4; ++g){
        int nb = g*4 + j;
        #pragma unroll
        for (int kb = 0; kb < 4; ++kb){
          v16h b = *(const v16h*)(wihfrag_f + (size_t)(kb*16 + nb)*512 + lane*16);
          acc[g] = wmma16(ain[kb], b, acc[g]);
        }
        v16h b0 = *(const v16h*)(whhfrag_f + (size_t)(nb)*512      + lane*16);
        acc[g] = wmma16(ah0, b0, acc[g]);
        v16h b1 = *(const v16h*)(whhfrag_f + (size_t)(16 + nb)*512 + lane*16);
        acc[g] = wmma16(ah1, b1, acc[g]);
      }
      #pragma unroll
      for (int r = 0; r < 8; ++r){
        float iv = fast_sig(acc[0][r]);
        float fv = fast_sig(acc[1][r]);
        float gv = fast_tanh(acc[2][r]);
        float ov = fast_sig(acc[3][r]);
        float cv = fv*c_st[j][r] + iv*gv;
        c_st[j][r] = cv;
        float hv = ov*fast_tanh(cv);
        int m = r + mrow_hi;
        int n = j*16 + ncol;
        st[m*64 + n] = (_Float16)hv;
        if (t == TSEQ-1) store_afrag_elem(obase, n, m, hv);  // dir f: k 0..63
      }
    }
    asm volatile("s_wait_dscnt 0" ::: "memory");
  }

  // ---- backward direction: single step at t=T-1, h0=c0=0 (no w_hh) ----
  {
    v16h ain[4];
    #pragma unroll
    for (int kb = 0; kb < 4; ++kb)
      ain[kb] = *(const v16h*)(h0frag + ((size_t)(tile*TSEQ + (TSEQ-1))*4 + kb)*512
                               + lane*16);
    #pragma unroll
    for (int j = 0; j < 4; ++j){
      v8f acc[4];
      #pragma unroll
      for (int g = 0; g < 4; ++g){
        int col = g*64 + j*16 + ncol;
        float binit = bih_b[col] + bhh_b[col];
        #pragma unroll
        for (int r = 0; r < 8; ++r) acc[g][r] = binit;
      }
      #pragma unroll
      for (int g = 0; g < 4; ++g){
        int nb = g*4 + j;
        #pragma unroll
        for (int kb = 0; kb < 4; ++kb){
          v16h b = *(const v16h*)(wihfrag_b + (size_t)(kb*16 + nb)*512 + lane*16);
          acc[g] = wmma16(ain[kb], b, acc[g]);
        }
      }
      #pragma unroll
      for (int r = 0; r < 8; ++r){
        float iv = fast_sig(acc[0][r]);
        float gv = fast_tanh(acc[2][r]);
        float ov = fast_sig(acc[3][r]);
        float cv = iv*gv;               // f*c0 = 0
        float hv = ov*fast_tanh(cv);
        store_afrag_elem(obase, 64 + j*16 + ncol, r + mrow_hi, hv);
      }
    }
  }
}

// ---------------------------------------------------------------------------
// GAT-1 projection: hW1 = hfin(16x128 f16) @ W1(128x24, padded to 32) via WMMA.
// ---------------------------------------------------------------------------
__global__ void __launch_bounds__(256)
proj1_k(const _Float16* __restrict__ hfinfrag,
        const _Float16* __restrict__ w1frag,
        float* __restrict__ hW){
  int w = threadIdx.x >> 5, lane = threadIdx.x & 31;
  int tile = blockIdx.x*8 + w;
  if (tile >= NTILES) return;
  v16h a[4];
  #pragma unroll
  for (int kb = 0; kb < 4; ++kb)
    a[kb] = *(const v16h*)(hfinfrag + ((size_t)tile*4 + kb)*512 + lane*16);
  v8f acc[2];
  #pragma unroll
  for (int nb = 0; nb < 2; ++nb)
    #pragma unroll
    for (int r = 0; r < 8; ++r) acc[nb][r] = 0.f;
  #pragma unroll
  for (int nb = 0; nb < 2; ++nb)
    #pragma unroll
    for (int kb = 0; kb < 4; ++kb){
      v16h b = *(const v16h*)(w1frag + (size_t)(kb*2 + nb)*512 + lane*16);
      acc[nb] = wmma16(a[kb], b, acc[nb]);
    }
  int base = tile*16;
  int mrow_hi = (lane >> 4) << 3;
  #pragma unroll
  for (int nb = 0; nb < 2; ++nb)
    #pragma unroll
    for (int r = 0; r < 8; ++r){
      int n = nb*16 + (lane & 15);
      if (n < FDIM) hW[(size_t)(base + r + mrow_hi)*FDIM + n] = acc[nb][r];
    }
}

// Per-node attention logits e_src/e_dst from projected features.
__global__ void node_att_k(const float* __restrict__ hW,
                           const float* __restrict__ a_src,
                           const float* __restrict__ a_dst,
                           float* __restrict__ es, float* __restrict__ ed){
  int n = blockIdx.x*blockDim.x + threadIdx.x;
  if (n >= NNODES) return;
  #pragma unroll
  for (int h = 0; h < HEADS; ++h){
    float s = 0.f, d = 0.f;
    #pragma unroll
    for (int c = 0; c < 4; ++c){
      float hv = hW[(size_t)n*FDIM + h*4 + c];
      s += hv*a_src[h*4 + c];
      d += hv*a_dst[h*4 + c];
    }
    es[(size_t)n*HEADS + h] = s;
    ed[(size_t)n*HEADS + h] = d;
  }
}

__global__ void init_att_k(unsigned* __restrict__ maxu,
                           float* __restrict__ denom, float* __restrict__ num){
  int i = blockIdx.x*blockDim.x + threadIdx.x;
  if (i < NNODES*FDIM) num[i] = 0.f;
  if (i < NNODES*HEADS){ denom[i] = 0.f; maxu[i] = 0u; }
}

// Pass 1: per-destination max of leaky_relu(e_src[s]+e_dst[d]).
__global__ void edge_max_k(const int* __restrict__ src, const int* __restrict__ dst,
                           int etot, const float* __restrict__ es,
                           const float* __restrict__ ed,
                           unsigned* __restrict__ maxu){
  int e = blockIdx.x*blockDim.x + threadIdx.x;
  if (e >= etot) return;
  int s, d;
  if (e < NEDGES){ s = src[e]; d = dst[e]; } else { s = e - NEDGES; d = s; }
  #pragma unroll
  for (int h = 0; h < HEADS; ++h){
    float v = es[(size_t)s*HEADS + h] + ed[(size_t)d*HEADS + h];
    v = v > 0.f ? v : 0.2f*v;
    atomicMax(&maxu[(size_t)d*HEADS + h], fenc(v));
  }
}

// Pass 2: accumulate exp-sums and exp-weighted feature numerators.
__global__ void edge_acc_k(const int* __restrict__ src, const int* __restrict__ dst,
                           int etot, const float* __restrict__ es,
                           const float* __restrict__ ed,
                           const unsigned* __restrict__ maxu,
                           float* __restrict__ denom, float* __restrict__ num,
                           const float* __restrict__ hW){
  int e = blockIdx.x*blockDim.x + threadIdx.x;
  if (e >= etot) return;
  int s, d;
  if (e < NEDGES){ s = src[e]; d = dst[e]; } else { s = e - NEDGES; d = s; }
  #pragma unroll
  for (int h = 0; h < HEADS; ++h){
    float v = es[(size_t)s*HEADS + h] + ed[(size_t)d*HEADS + h];
    v = v > 0.f ? v : 0.2f*v;
    float ex = fast_exp(v - fdec(maxu[(size_t)d*HEADS + h]));
    atomicAdd(&denom[(size_t)d*HEADS + h], ex);
    #pragma unroll
    for (int c = 0; c < 4; ++c)
      atomicAdd(&num[(size_t)d*FDIM + h*4 + c],
                ex*hW[(size_t)s*FDIM + h*4 + c]);
  }
}

// Pass 3 (GAT-1): h1 = elu(num/denom + b1)
__global__ void node_fin1_k(const float* __restrict__ num,
                            const float* __restrict__ denom,
                            const float* __restrict__ b1,
                            float* __restrict__ h1){
  int n = blockIdx.x*blockDim.x + threadIdx.x;
  if (n >= NNODES) return;
  #pragma unroll
  for (int k = 0; k < FDIM; ++k){
    float v = num[(size_t)n*FDIM + k]*fast_rcp(denom[(size_t)n*HEADS + (k >> 2)])
              + b1[k];
    h1[(size_t)n*FDIM + k] = v > 0.f ? v : (fast_exp(v) - 1.f);
  }
}

// GAT-2 projection (24x24, VALU; LDS-staged weights) + logits.
__global__ void proj2_k(const float* __restrict__ h1, const float* __restrict__ W2,
                        const float* __restrict__ a_src,
                        const float* __restrict__ a_dst,
                        float* __restrict__ h2W,
                        float* __restrict__ es, float* __restrict__ ed){
  __shared__ float sW[FDIM*FDIM];
  __shared__ float sas[FDIM], sad[FDIM];
  int tid = threadIdx.x;
  for (int i = tid; i < FDIM*FDIM; i += blockDim.x) sW[i] = W2[i];
  if (tid < FDIM){ sas[tid] = a_src[tid]; sad[tid] = a_dst[tid]; }
  __syncthreads();
  int n = blockIdx.x*blockDim.x + tid;
  if (n >= NNODES) return;
  float hv[FDIM];
  #pragma unroll
  for (int k = 0; k < FDIM; ++k) hv[k] = h1[(size_t)n*FDIM + k];
  float out[FDIM];
  #pragma unroll
  for (int o = 0; o < FDIM; ++o){
    float acc = 0.f;
    #pragma unroll
    for (int k = 0; k < FDIM; ++k) acc += hv[k]*sW[k*FDIM + o];
    out[o] = acc;
    h2W[(size_t)n*FDIM + o] = acc;
  }
  #pragma unroll
  for (int h = 0; h < HEADS; ++h){
    float s = 0.f, d = 0.f;
    #pragma unroll
    for (int c = 0; c < 4; ++c){
      s += out[h*4 + c]*sas[h*4 + c];
      d += out[h*4 + c]*sad[h*4 + c];
    }
    es[(size_t)n*HEADS + h] = s;
    ed[(size_t)n*HEADS + h] = d;
  }
}

// Pass 3 (GAT-2) fused with final linear: out = elu(num/den + b2) @ Wlin + blin
__global__ void node_fin2_k(const float* __restrict__ num,
                            const float* __restrict__ denom,
                            const float* __restrict__ b2,
                            const float* __restrict__ Wlin,
                            const float* __restrict__ blin,
                            float* __restrict__ out){
  int n = blockIdx.x*blockDim.x + threadIdx.x;
  if (n >= NNODES) return;
  float acc = blin[0];
  #pragma unroll
  for (int k = 0; k < FDIM; ++k){
    float v = num[(size_t)n*FDIM + k]*fast_rcp(denom[(size_t)n*HEADS + (k >> 2)])
              + b2[k];
    v = v > 0.f ? v : (fast_exp(v) - 1.f);
    acc += v*Wlin[k];
  }
  out[n] = acc;
}

// ---------------------------------------------------------------------------
extern "C" void kernel_launch(void* const* d_in, const int* in_sizes, int n_in,
                              void* d_out, int out_size, void* d_ws, size_t ws_size,
                              hipStream_t stream){
  (void)in_sizes; (void)n_in; (void)out_size; (void)ws_size;
  const float* x    = (const float*)d_in[0];
  const int*   eidx = (const int*)  d_in[1];
  const int*   e_src = eidx;
  const int*   e_dst = eidx + NEDGES;
  const float* w_ih_l0f = (const float*)d_in[2];
  const float* w_hh_l0f = (const float*)d_in[3];
  const float* b_ih_l0f = (const float*)d_in[4];
  const float* b_hh_l0f = (const float*)d_in[5];
  const float* w_ih_l0b = (const float*)d_in[6];
  const float* w_hh_l0b = (const float*)d_in[7];
  const float* b_ih_l0b = (const float*)d_in[8];
  const float* b_hh_l0b = (const float*)d_in[9];
  const float* w_ih_l1f = (const float*)d_in[10];
  const float* w_hh_l1f = (const float*)d_in[11];
  const float* b_ih_l1f = (const float*)d_in[12];
  const float* b_hh_l1f = (const float*)d_in[13];
  const float* w_ih_l1b = (const float*)d_in[14];
  const float* b_ih_l1b = (const float*)d_in[16];
  const float* b_hh_l1b = (const float*)d_in[17];
  const float* W1     = (const float*)d_in[18];
  const float* a1_src = (const float*)d_in[19];
  const float* a1_dst = (const float*)d_in[20];
  const float* b1     = (const float*)d_in[21];
  const float* W2     = (const float*)d_in[22];
  const float* a2_src = (const float*)d_in[23];
  const float* a2_dst = (const float*)d_in[24];
  const float* b2     = (const float*)d_in[25];
  const float* Wlin   = (const float*)d_in[26];
  const float* blin   = (const float*)d_in[27];

  // ---- workspace carve-up ----
  char* ws = (char*)d_ws;
  size_t off = 0;
  auto take = [&](size_t bytes)->char*{
    char* p = ws + off;
    off += (bytes + 255) & ~(size_t)255;
    return p;
  };
  _Float16* h0frag   = (_Float16*)take((size_t)NTILES*TSEQ*2048*2);
  _Float16* hfinfrag = (_Float16*)take((size_t)NTILES*2048*2);
  _Float16* whh0f = (_Float16*)take((size_t)32*512*2);
  _Float16* whh0b = (_Float16*)take((size_t)32*512*2);
  _Float16* wih1f = (_Float16*)take((size_t)64*512*2);
  _Float16* wih1b = (_Float16*)take((size_t)64*512*2);
  _Float16* whh1f = (_Float16*)take((size_t)32*512*2);
  _Float16* w1frg = (_Float16*)take((size_t)8*512*2);
  float*    hW1   = (float*)take((size_t)NNODES*FDIM*4);
  float*    es1   = (float*)take((size_t)NNODES*HEADS*4);
  float*    ed1   = (float*)take((size_t)NNODES*HEADS*4);
  unsigned* maxu  = (unsigned*)take((size_t)NNODES*HEADS*4);
  float*    denom = (float*)take((size_t)NNODES*HEADS*4);
  float*    num   = (float*)take((size_t)NNODES*FDIM*4);
  float*    h1    = (float*)take((size_t)NNODES*FDIM*4);
  float*    h2W   = (float*)take((size_t)NNODES*FDIM*4);
  float*    es2   = (float*)take((size_t)NNODES*HEADS*4);
  float*    ed2   = (float*)take((size_t)NNODES*HEADS*4);

  // ---- pack weights into WMMA B-fragment order ----
  pack_bfrag_k<<<64, 256, 0, stream>>>(w_hh_l0f, whh0f, 16, 2, 256, 64, 1);
  pack_bfrag_k<<<64, 256, 0, stream>>>(w_hh_l0b, whh0b, 16, 2, 256, 64, 1);
  pack_bfrag_k<<<128,256, 0, stream>>>(w_ih_l1f, wih1f, 16, 4, 256,128, 1);
  pack_bfrag_k<<<128,256, 0, stream>>>(w_ih_l1b, wih1b, 16, 4, 256,128, 1);
  pack_bfrag_k<<<64, 256, 0, stream>>>(w_hh_l1f, whh1f, 16, 2, 256, 64, 1);
  pack_bfrag_k<<<16, 256, 0, stream>>>(W1,       w1frg,  2, 4, FDIM,128, 0);

  // ---- dense pipeline (WMMA) ----
  int lstm_grid = (NTILES + 7)/8;
  lstm0_k<<<lstm_grid, 256, 0, stream>>>(x, w_ih_l0f, w_ih_l0b,
                                         b_ih_l0f, b_hh_l0f, b_ih_l0b, b_hh_l0b,
                                         whh0f, whh0b, h0frag);
  lstm1_k<<<lstm_grid, 256, 0, stream>>>(h0frag,
                                         b_ih_l1f, b_hh_l1f, b_ih_l1b, b_hh_l1b,
                                         wih1f, whh1f, wih1b, hfinfrag);
  proj1_k<<<lstm_grid, 256, 0, stream>>>(hfinfrag, w1frg, hW1);

  // ---- GAT layer 1 ----
  int ngrid = (NNODES + 255)/256;
  int igrid = (NNODES*FDIM + 255)/256;
  int etot  = NEDGES + NNODES;
  int egrid = (etot + 255)/256;
  node_att_k<<<ngrid, 256, 0, stream>>>(hW1, a1_src, a1_dst, es1, ed1);
  init_att_k<<<igrid, 256, 0, stream>>>(maxu, denom, num);
  edge_max_k<<<egrid, 256, 0, stream>>>(e_src, e_dst, etot, es1, ed1, maxu);
  edge_acc_k<<<egrid, 256, 0, stream>>>(e_src, e_dst, etot, es1, ed1,
                                        maxu, denom, num, hW1);
  node_fin1_k<<<ngrid, 256, 0, stream>>>(num, denom, b1, h1);

  // ---- GAT layer 2 + final linear ----
  proj2_k<<<ngrid, 256, 0, stream>>>(h1, W2, a2_src, a2_dst, h2W, es2, ed2);
  init_att_k<<<igrid, 256, 0, stream>>>(maxu, denom, num);
  edge_max_k<<<egrid, 256, 0, stream>>>(e_src, e_dst, etot, es2, ed2, maxu);
  edge_acc_k<<<egrid, 256, 0, stream>>>(e_src, e_dst, etot, es2, ed2,
                                        maxu, denom, num, h2W);
  node_fin2_k<<<ngrid, 256, 0, stream>>>(num, denom, b2, Wlin, blin,
                                         (float*)d_out);
}
